// LSTM_24988119728250
// MI455X (gfx1250) — matrix operand order repaired and checked by reading, not compiled, not run on hardware
//
#include <hip/hip_runtime.h>
#include <hip/hip_bf16.h>

// ---------- types ----------
typedef __attribute__((ext_vector_type(16))) __bf16 v16bf;
typedef __attribute__((ext_vector_type(8)))  __bf16 v8bf;
typedef __attribute__((ext_vector_type(4)))  __bf16 v4bf;
typedef __attribute__((ext_vector_type(8)))  float  v8f;

union FragBF { v16bf v; v8bf h[2]; };

__device__ __forceinline__ float sigmoidf_(float x) {
    return 1.0f / (1.0f + __expf(-x));
}

// ---------- constants ----------
#define BB 64
#define SS 512
#define II 1024
#define HH 1024
#define G4H 4096

// ---------- f32 -> bf16 convert, 4 elements/lane ----------
__global__ __launch_bounds__(256) void cvt_f32_bf16_x4(const float* __restrict__ src,
                                                       __bf16* __restrict__ dst, int n4) {
    int i = blockIdx.x * 256 + threadIdx.x;
    if (i < n4) {
        float4 f = ((const float4*)src)[i];
        v4bf o;
        o[0] = (__bf16)f.x; o[1] = (__bf16)f.y; o[2] = (__bf16)f.z; o[3] = (__bf16)f.w;
        ((v4bf*)dst)[i] = o;
    }
}

// ---------- big input-projection GEMM: G[r, n] = sum_k A[r, k] * W[n, k] ----------
// Wave tile: 32 rows x 64 cols (2 A-frags x 4 B-frags = 8 accumulators), K = 1024,
// software-pipelined (preload k+32 while WMMAs for k issue).
// A rows r = t*64 + b.  xlayout==0: A element at (b*512 + t)*1024 + k  (x tensor [B,S,I])
//                       xlayout==1: A element at r*1024 + k           (h0_seq [S,B,H])
__global__ __launch_bounds__(256) void gemm_inproj(const __bf16* __restrict__ A,
                                                   const __bf16* __restrict__ W,
                                                   float* __restrict__ G,
                                                   int xlayout) {
    int wid  = (blockIdx.x << 3) + (threadIdx.x >> 5);   // global wave id
    int lane = threadIdx.x & 31;
    int hi = lane >> 4, lo = lane & 15;
    int nsuper = wid & 63;          // 64 N-supertiles of 64 cols
    int mtile  = wid >> 6;          // 1024 M-tiles of 32 rows

    const __bf16* arow[2];
#pragma unroll
    for (int mi = 0; mi < 2; ++mi) {
        int r = mtile * 32 + mi * 16 + lo;
        if (xlayout == 0) {
            int b = r & 63, t = r >> 6;
            arow[mi] = A + (size_t)(b * SS + t) * II;
        } else {
            arow[mi] = A + (size_t)r * HH;
        }
    }
    const __bf16* wrow[4];
#pragma unroll
    for (int s = 0; s < 4; ++s)
        wrow[s] = W + (size_t)(nsuper * 64 + s * 16 + lo) * 1024;

    v8f acc[8] = {};
    FragBF aC[2], bC[4], aN[2], bN[4];
#pragma unroll
    for (int mi = 0; mi < 2; ++mi) {
        aC[mi].h[0] = *(const v8bf*)(arow[mi] + hi * 16);
        aC[mi].h[1] = *(const v8bf*)(arow[mi] + hi * 16 + 8);
    }
#pragma unroll
    for (int s = 0; s < 4; ++s) {
        bC[s].h[0] = *(const v8bf*)(wrow[s] + hi * 16);
        bC[s].h[1] = *(const v8bf*)(wrow[s] + hi * 16 + 8);
    }
    for (int k0 = 32; k0 < 1024; k0 += 32) {
#pragma unroll
        for (int mi = 0; mi < 2; ++mi) {
            aN[mi].h[0] = *(const v8bf*)(arow[mi] + k0 + hi * 16);
            aN[mi].h[1] = *(const v8bf*)(arow[mi] + k0 + hi * 16 + 8);
        }
#pragma unroll
        for (int s = 0; s < 4; ++s) {
            bN[s].h[0] = *(const v8bf*)(wrow[s] + k0 + hi * 16);
            bN[s].h[1] = *(const v8bf*)(wrow[s] + k0 + hi * 16 + 8);
        }
#pragma unroll
        for (int mi = 0; mi < 2; ++mi)
#pragma unroll
            for (int s = 0; s < 4; ++s)
                acc[mi * 4 + s] = __builtin_amdgcn_wmma_f32_16x16x32_bf16(
                    false, aC[mi].v, false, bC[s].v, (short)0, acc[mi * 4 + s], false, false);
#pragma unroll
        for (int mi = 0; mi < 2; ++mi) aC[mi] = aN[mi];
#pragma unroll
        for (int s = 0; s < 4; ++s) bC[s] = bN[s];
    }
#pragma unroll
    for (int mi = 0; mi < 2; ++mi)
#pragma unroll
        for (int s = 0; s < 4; ++s)
            acc[mi * 4 + s] = __builtin_amdgcn_wmma_f32_16x16x32_bf16(
                false, aC[mi].v, false, bC[s].v, (short)0, acc[mi * 4 + s], false, false);

#pragma unroll
    for (int mi = 0; mi < 2; ++mi)
#pragma unroll
        for (int s = 0; s < 4; ++s) {
            int n = nsuper * 64 + s * 16 + lo;
#pragma unroll
            for (int e = 0; e < 8; ++e) {
                int m = mtile * 32 + mi * 16 + e + hi * 8;
                G[(size_t)m * G4H + n] = acc[mi * 4 + s][e];
            }
        }
}

// ---------- persistent recurrence kernel: all 512 timesteps in one launch ----------
// 32 blocks x 256 threads = 256 waves; wave owns one 16x16 (b, j) tile for all 4 gates,
// for every t. Device-scope barrier (per-step counter + s_sleep spin) between steps.
__global__ __launch_bounds__(256) void lstm_recurrence(const float* __restrict__ Gx0,   // [512][64][4096]
                                                       const __bf16* __restrict__ Wh,   // [4096][1024]
                                                       const float* __restrict__ bias,  // [4096]
                                                       float* __restrict__ C,           // [64][1024]
                                                       const __bf16* __restrict__ Hzero,// [64][1024] zeros
                                                       __bf16* __restrict__ Hseq,       // seq or ping-pong
                                                       int hseq_mode,                   // 0: [512][64][1024], 1: [2][64][1024]
                                                       float* __restrict__ OutF32,      // layer1 outputs base or null
                                                       float* __restrict__ HLast,
                                                       float* __restrict__ CLast,
                                                       int* __restrict__ Bar,           // [512]
                                                       int nblocks) {
    int wid  = (blockIdx.x << 3) + (threadIdx.x >> 5);   // 0..255
    int lane = threadIdx.x & 31;
    int hi = lane >> 4, lo = lane & 15;
    int mtile = wid & 3;            // 4 tiles of 16 batch rows
    int jtile = wid >> 2;           // 64 tiles of 16 hidden cols
    int jg = jtile * 16 + lo;

    const __bf16* wrow[4];
#pragma unroll
    for (int g = 0; g < 4; ++g)
        wrow[g] = Wh + (size_t)(g * HH + jtile * 16 + lo) * HH;
    float b_i = bias[jg], b_f = bias[HH + jg], b_g = bias[2 * HH + jg], b_o = bias[3 * HH + jg];

    for (int t = 0; t < SS; ++t) {
        const __bf16* hin = (t == 0) ? Hzero
                          : (hseq_mode == 0 ? Hseq + (size_t)(t - 1) * BB * HH
                                            : Hseq + (size_t)((t - 1) & 1) * BB * HH);
        __bf16* hout = (hseq_mode == 0) ? Hseq + (size_t)t * BB * HH
                                        : Hseq + (size_t)(t & 1) * BB * HH;
        const float* Gx = Gx0 + (size_t)t * BB * G4H;
        const __bf16* hrow = hin + (size_t)(mtile * 16 + lo) * HH;

        v8f acc[4] = {};
        FragBF aC, bCf[4], aN, bNf[4];
        aC.h[0] = *(const v8bf*)(hrow + hi * 16);
        aC.h[1] = *(const v8bf*)(hrow + hi * 16 + 8);
#pragma unroll
        for (int g = 0; g < 4; ++g) {
            bCf[g].h[0] = *(const v8bf*)(wrow[g] + hi * 16);
            bCf[g].h[1] = *(const v8bf*)(wrow[g] + hi * 16 + 8);
        }
        for (int k0 = 32; k0 < 1024; k0 += 32) {
            aN.h[0] = *(const v8bf*)(hrow + k0 + hi * 16);
            aN.h[1] = *(const v8bf*)(hrow + k0 + hi * 16 + 8);
#pragma unroll
            for (int g = 0; g < 4; ++g) {
                bNf[g].h[0] = *(const v8bf*)(wrow[g] + k0 + hi * 16);
                bNf[g].h[1] = *(const v8bf*)(wrow[g] + k0 + hi * 16 + 8);
            }
#pragma unroll
            for (int g = 0; g < 4; ++g)
                acc[g] = __builtin_amdgcn_wmma_f32_16x16x32_bf16(
                    false, aC.v, false, bCf[g].v, (short)0, acc[g], false, false);
            aC = aN;
#pragma unroll
            for (int g = 0; g < 4; ++g) bCf[g] = bNf[g];
        }
#pragma unroll
        for (int g = 0; g < 4; ++g)
            acc[g] = __builtin_amdgcn_wmma_f32_16x16x32_bf16(
                false, aC.v, false, bCf[g].v, (short)0, acc[g], false, false);

        // ---- fused LSTM pointwise (4 gates in-register) ----
#pragma unroll
        for (int e = 0; e < 8; ++e) {
            int b = mtile * 16 + e + hi * 8;
            const float* gxr = Gx + (size_t)b * G4H;
            float gi = acc[0][e] + gxr[jg]          + b_i;
            float gf = acc[1][e] + gxr[HH + jg]     + b_f;
            float gg = acc[2][e] + gxr[2 * HH + jg] + b_g;
            float go = acc[3][e] + gxr[3 * HH + jg] + b_o;
            float ig = sigmoidf_(gi);
            float fg = sigmoidf_(gf);
            float og = sigmoidf_(go);
            float cg = tanhf(gg);
            size_t ci = (size_t)b * HH + jg;
            float cn = fg * C[ci] + ig * cg;
            C[ci] = cn;
            float hn = og * tanhf(cn);
            hout[ci] = (__bf16)hn;
            if (OutF32) OutF32[((size_t)b * SS + t) * HH + jg] = hn;
            if (HLast) { HLast[ci] = hn; CLast[ci] = cn; }
        }

        // ---- device-scope step barrier ----
        __threadfence();                 // release this wave's h/c stores
        __syncthreads();
        if (threadIdx.x == 0) {
            atomicAdd(Bar + t, 1);
            while (__hip_atomic_load(Bar + t, __ATOMIC_ACQUIRE, __HIP_MEMORY_SCOPE_AGENT) < nblocks)
                __builtin_amdgcn_s_sleep(2);
        }
        __syncthreads();
    }
}

// ---------- workspace layout (bytes) ----------
#define OFF_WX0B   ((size_t)0)
#define OFF_WH0B   (OFF_WX0B + (size_t)G4H * II * 2)        //  8 MB
#define OFF_WX1B   (OFF_WH0B + (size_t)G4H * HH * 2)        //  8 MB
#define OFF_WH1B   (OFF_WX1B + (size_t)G4H * HH * 2)        //  8 MB
#define OFF_XB     (OFF_WH1B + (size_t)G4H * HH * 2)        //  8 MB
#define OFF_H0SEQ  (OFF_XB   + (size_t)BB * SS * II * 2)    // 64 MB
#define OFF_GATES  (OFF_H0SEQ + (size_t)SS * BB * HH * 2)   // 64 MB
#define OFF_H1BUF  (OFF_GATES + (size_t)SS * BB * G4H * 4)  // 512 MB
#define OFF_CCUR   (OFF_H1BUF + (size_t)2 * BB * HH * 2)
#define OFF_ZEROH  (OFF_CCUR  + (size_t)BB * HH * 4)
#define OFF_BAR    (OFF_ZEROH + (size_t)BB * HH * 2)        // 2 x 512 ints

extern "C" void kernel_launch(void* const* d_in, const int* in_sizes, int n_in,
                              void* d_out, int out_size, void* d_ws, size_t ws_size,
                              hipStream_t stream) {
    const float* x   = (const float*)d_in[0];
    const float* Wx0 = (const float*)d_in[1];
    const float* Wh0 = (const float*)d_in[2];
    const float* bh0 = (const float*)d_in[3];
    const float* Wx1 = (const float*)d_in[4];
    const float* Wh1 = (const float*)d_in[5];
    const float* bh1 = (const float*)d_in[6];

    char* ws = (char*)d_ws;
    __bf16* wx0b  = (__bf16*)(ws + OFF_WX0B);
    __bf16* wh0b  = (__bf16*)(ws + OFF_WH0B);
    __bf16* wx1b  = (__bf16*)(ws + OFF_WX1B);
    __bf16* wh1b  = (__bf16*)(ws + OFF_WH1B);
    __bf16* xb    = (__bf16*)(ws + OFF_XB);
    __bf16* h0seq = (__bf16*)(ws + OFF_H0SEQ);
    float*  gates = (float*)(ws + OFF_GATES);
    __bf16* h1buf = (__bf16*)(ws + OFF_H1BUF);
    float*  ccur  = (float*)(ws + OFF_CCUR);
    __bf16* zeroh = (__bf16*)(ws + OFF_ZEROH);
    int*    bar0  = (int*)(ws + OFF_BAR);
    int*    bar1  = bar0 + SS;

    const int WN4 = (G4H * II) / 4;          // weight elements / 4
    const int XN4 = (BB * SS * II) / 4;
    cvt_f32_bf16_x4<<<(WN4 + 255) / 256, 256, 0, stream>>>(Wx0, wx0b, WN4);
    cvt_f32_bf16_x4<<<(WN4 + 255) / 256, 256, 0, stream>>>(Wh0, wh0b, WN4);
    cvt_f32_bf16_x4<<<(WN4 + 255) / 256, 256, 0, stream>>>(Wx1, wx1b, WN4);
    cvt_f32_bf16_x4<<<(WN4 + 255) / 256, 256, 0, stream>>>(Wh1, wh1b, WN4);
    cvt_f32_bf16_x4<<<(XN4 + 255) / 256, 256, 0, stream>>>(x, xb, XN4);

    // ---- layer 0 ----
    gemm_inproj<<<8192, 256, 0, stream>>>(xb, wx0b, gates, /*xlayout=*/0);
    hipMemsetAsync(zeroh, 0, (size_t)BB * HH * 2, stream);
    hipMemsetAsync(ccur,  0, (size_t)BB * HH * 4, stream);
    hipMemsetAsync(bar0,  0, (size_t)SS * 4, stream);
    lstm_recurrence<<<32, 256, 0, stream>>>(gates, wh0b, bh0, ccur, zeroh,
                                            h0seq, /*hseq_mode=*/0,
                                            nullptr, nullptr, nullptr, bar0, 32);

    // ---- layer 1 ----
    gemm_inproj<<<8192, 256, 0, stream>>>(h0seq, wx1b, gates, /*xlayout=*/1);
    hipMemsetAsync(h1buf, 0, (size_t)BB * HH * 2, stream);   // ping-pong slot 0
    hipMemsetAsync(ccur,  0, (size_t)BB * HH * 4, stream);
    hipMemsetAsync(bar1,  0, (size_t)SS * 4, stream);

    float* out   = (float*)d_out;
    float* hlast = out + (size_t)BB * SS * HH;
    float* clast = hlast + (size_t)BB * HH;
    lstm_recurrence<<<32, 256, 0, stream>>>(gates, wh1b, bh1, ccur, zeroh,
                                            h1buf, /*hseq_mode=*/1,
                                            out, hlast, clast, bar1, 32);
}
